// CompoundProteinInteractionPrediction_87634512708081
// MI455X (gfx1250) — compile-verified, matrix operand order checked by reading, not run on hardware
//
#include <hip/hip_runtime.h>
#include <hip/hip_bf16.h>
#include <math.h>

// ---------------------------------------------------------------------------
// CompoundProteinInteractionPrediction for MI455X (gfx1250, wave32, WMMA)
// Dominant op: adjacency(8192x8192, binary) @ hs(8192x128)  x3 layers.
// adjacency -> fp16 once (exact for 0/1); all dense linears on
// v_wmma_f32_16x16x32_f16 with fp32 accumulate; hs stored transposed fp16 so
// A and B fragments are contiguous 16B per-lane loads. Conv tile staging uses
// global_load_async_to_lds_b128 (ASYNCcnt) per CDNA5 ISA.
// ---------------------------------------------------------------------------

#define DIMC     128
#define NATOMS   8192
#define LPROT    8192
#define WIN      11
#define KS       23          // 2*WIN+1
#define KS2      (KS*KS)     // 529

typedef __attribute__((ext_vector_type(16))) _Float16 v16h;
typedef __attribute__((ext_vector_type(8)))  _Float16 v8h;
typedef __attribute__((ext_vector_type(4)))  _Float16 v4h;
typedef __attribute__((ext_vector_type(8)))  float    v8f;

// ---- WMMA fragment loaders (CDNA5 ISA 7.12.2 layouts, wave32) --------------
// Lane g=lane>>4, r=lane&15 holds K = g*8 + {0..7} (VGPR0-3) and
// 16 + g*8 + {0..7} (VGPR4-7): two contiguous 16-byte loads per lane.
// Same mapping loads a B 32x16 fragment from [N][K] row-major storage.
__device__ __forceinline__ v16h frag_rm_f16(const _Float16* __restrict__ A,
                                            int r0, int k0, int ld) {
  const int lane = threadIdx.x & 31;
  const _Float16* p = A + (size_t)(r0 + (lane & 15)) * ld
                        + (size_t)(k0 + ((lane >> 4) << 3));
  v8h lo = *(const v8h*)p;
  v8h hi = *(const v8h*)(p + 16);
  v16h r;
#pragma unroll
  for (int i = 0; i < 8; ++i) { r[i] = lo[i]; r[8 + i] = hi[i]; }
  return r;
}

// Same fragment but source is fp32 (converted in-register to fp16).
__device__ __forceinline__ v16h frag_rm_f32(const float* __restrict__ A,
                                            int r0, int k0, int ld) {
  const int lane = threadIdx.x & 31;
  const float* p = A + (size_t)(r0 + (lane & 15)) * ld
                     + (size_t)(k0 + ((lane >> 4) << 3));
  v16h r;
#pragma unroll
  for (int i = 0; i < 8; ++i) {
    r[i]     = (_Float16)p[i];
    r[8 + i] = (_Float16)p[16 + i];
  }
  return r;
}

// ---- elementwise / gather kernels ------------------------------------------
__global__ void k_adj_to_half(const float4* __restrict__ in,
                              v4h* __restrict__ out, int n4) {
  int i = blockIdx.x * blockDim.x + threadIdx.x;
  if (i >= n4) return;
  float4 v = in[i];
  v4h h; h[0] = (_Float16)v.x; h[1] = (_Float16)v.y;
         h[2] = (_Float16)v.z; h[3] = (_Float16)v.w;
  out[i] = h;
}

__global__ void k_gather_f32h(const int* __restrict__ idx,
                              const float* __restrict__ emb,
                              float* __restrict__ xs,
                              _Float16* __restrict__ xsH) {
  int i = blockIdx.x * blockDim.x + threadIdx.x;   // over NATOMS*DIMC
  int r = i >> 7, c = i & 127;
  float v = emb[(size_t)idx[r] * DIMC + c];
  xs[i] = v;
  xsH[i] = (_Float16)v;
}

__global__ void k_gather_f32(const int* __restrict__ idx,
                             const float* __restrict__ emb,
                             float* __restrict__ xs) {
  int i = blockIdx.x * blockDim.x + threadIdx.x;
  int r = i >> 7, c = i & 127;
  xs[i] = emb[(size_t)idx[r] * DIMC + c];
}

// ---- GEMM 1: hsT = relu(xsH @ Wg^T + bg), output fp16 TRANSPOSED [128][8192]
// grid = M/16 blocks, 256 threads = 8 waves; wave w owns N-tile n0 = 16*w.
__global__ void k_lin_to_hsT(const _Float16* __restrict__ xsH,
                             const float* __restrict__ Wg,   // [128][128]
                             const float* __restrict__ bg,   // [128]
                             _Float16* __restrict__ hsT) {   // [128][8192]
  const int m0 = blockIdx.x * 16;
  const int n0 = (threadIdx.x >> 5) * 16;
  v8f acc = {};
#pragma unroll
  for (int k0 = 0; k0 < DIMC; k0 += 32) {
    v16h a = frag_rm_f16(xsH, m0, k0, DIMC);
    v16h b = frag_rm_f32(Wg, n0, k0, DIMC);     // B col n = Wg row n
    acc = __builtin_amdgcn_wmma_f32_16x16x32_f16(false, a, false, b,
                                                 (short)0, acc, false, false);
  }
  const int lane = threadIdx.x & 31;
  const int g = lane >> 4, n = n0 + (lane & 15);
  const float bias = bg[n];
  v8h h;
#pragma unroll
  for (int r = 0; r < 8; ++r) h[r] = (_Float16)fmaxf(acc[r] + bias, 0.0f);
  *(v8h*)(hsT + (size_t)n * NATOMS + m0 + g * 8) = h;   // transposed store
}

// ---- GEMM 2 (dominant): xs += adjH @ hs (as hsT); refresh xsH --------------
// Register-blocked: wave owns 1 M-tile x 2 N-tiles -> one A fragment feeds two
// WMMAs (A = the 128MB streamed operand). Block = 8 waves (2 wrows x 4 wcols),
// tile 32x128, grid NATOMS/32 = 256 blocks.
__global__ void k_adj_gemm(const _Float16* __restrict__ adjH, // [8192][8192]
                           const _Float16* __restrict__ hsT,  // [128][8192]
                           float* __restrict__ xs,            // [8192][128]
                           _Float16* __restrict__ xsH) {
  const int wave = threadIdx.x >> 5;
  const int m0 = blockIdx.x * 32 + (wave >> 2) * 16;
  const int n0 = (wave & 3) * 32;               // covers n0 and n0+16
  v8f acc0 = {}, acc1 = {};
#pragma unroll 2
  for (int k0 = 0; k0 < NATOMS; k0 += 32) {     // 256 k-steps
    v16h a  = frag_rm_f16(adjH, m0, k0, NATOMS);
    v16h b0 = frag_rm_f16(hsT, n0,      k0, NATOMS);
    v16h b1 = frag_rm_f16(hsT, n0 + 16, k0, NATOMS);
    acc0 = __builtin_amdgcn_wmma_f32_16x16x32_f16(false, a, false, b0,
                                                  (short)0, acc0, false, false);
    acc1 = __builtin_amdgcn_wmma_f32_16x16x32_f16(false, a, false, b1,
                                                  (short)0, acc1, false, false);
  }
  const int lane = threadIdx.x & 31;
  const int g = lane >> 4, nl = lane & 15;
  const int mb = m0 + g * 8;
#pragma unroll
  for (int r = 0; r < 8; ++r) {
    size_t i0 = (size_t)(mb + r) * DIMC + n0 + nl;
    float v0 = xs[i0] + acc0[r];
    xs[i0] = v0; xsH[i0] = (_Float16)v0;
    size_t i1 = (size_t)(mb + r) * DIMC + n0 + 16 + nl;
    float v1 = xs[i1] + acc1[r];
    xs[i1] = v1; xsH[i1] = (_Float16)v1;
  }
}

// ---- GEMM 3: hs2 = relu(convOut @ Wa^T + ba), fp32 in/out ------------------
__global__ void k_lin_f32(const float* __restrict__ A,     // [8192][128]
                          const float* __restrict__ Wa,    // [128][128]
                          const float* __restrict__ ba,
                          float* __restrict__ C) {         // [8192][128]
  const int m0 = blockIdx.x * 16;
  const int n0 = (threadIdx.x >> 5) * 16;
  v8f acc = {};
#pragma unroll
  for (int k0 = 0; k0 < DIMC; k0 += 32) {
    v16h a = frag_rm_f32(A,  m0, k0, DIMC);
    v16h b = frag_rm_f32(Wa, n0, k0, DIMC);
    acc = __builtin_amdgcn_wmma_f32_16x16x32_f16(false, a, false, b,
                                                 (short)0, acc, false, false);
  }
  const int lane = threadIdx.x & 31;
  const int g = lane >> 4, n = n0 + (lane & 15);
  const int mb = m0 + g * 8;
  const float bias = ba[n];
#pragma unroll
  for (int r = 0; r < 8; ++r)
    C[(size_t)(mb + r) * DIMC + n] = fmaxf(acc[r] + bias, 0.0f);
}

// ---- 23x23 "image" conv over [L,128] with zero pad 11 both dims, +bias,relu
// Tile staged in LDS; interior spans copied with GLOBAL_LOAD_ASYNC_TO_LDS_B128
// (ASYNCcnt); halo cells zeroed with plain DS stores (disjoint addresses, so
// the DScnt/ASYNCcnt split is race-free before the barrier).
#define CTILE 32
#define CROWS (CTILE + 2 * WIN)   // 54
#define CPADL 12                  // left pad (16B-aligned interior start)
#define CW    152                 // row pitch: 12 + 128 + 12
__global__ void k_conv(const float* __restrict__ in,   // [8192][128]
                       const float* __restrict__ w,    // [23][23]
                       const float* __restrict__ bptr, // scalar
                       float* __restrict__ out) {
  __shared__ __align__(16) float tile[CROWS * CW];
  __shared__ float wsm[KS2];
  const int l0 = blockIdx.x * CTILE;
  for (int i = threadIdx.x; i < KS2; i += blockDim.x) wsm[i] = w[i];
  // zero halo cells only (interior is filled by async copies)
  for (int i = threadIdx.x; i < CROWS * CW; i += blockDim.x) {
    int rr = i / CW, cc = i - rr * CW;
    int l = l0 + rr - WIN;
    bool interior = (l >= 0) && (l < LPROT) && (cc >= CPADL) &&
                    (cc < CPADL + DIMC);
    if (!interior) tile[i] = 0.0f;
  }
  // async-copy interior rows: 128 floats = 32 x b128 per row
  for (int t = threadIdx.x; t < CROWS * 32; t += blockDim.x) {
    int rr = t >> 5, seg = t & 31;
    int l = l0 + rr - WIN;
    if (l >= 0 && l < LPROT) {
      unsigned ldsoff = (unsigned)(size_t)&tile[rr * CW + CPADL + seg * 4];
      unsigned long long ga =
          (unsigned long long)(size_t)(in + (size_t)l * DIMC + seg * 4);
      asm volatile("global_load_async_to_lds_b128 %0, %1, off"
                   :: "v"(ldsoff), "v"(ga) : "memory");
    }
  }
  asm volatile("s_wait_asynccnt 0x0" ::: "memory");
  __syncthreads();
  const float bias = *bptr;
  for (int i = threadIdx.x; i < CTILE * DIMC; i += blockDim.x) {
    int rr = i >> 7, d = i & 127;
    float acc = bias;
#pragma unroll 1
    for (int kr = 0; kr < KS; ++kr) {
      // input col (d+kc-11) lives at tile col CPADL + d + kc - WIN = d+kc+1
      const float* trow = &tile[(rr + kr) * CW + d + 1];
      const float* wrow = &wsm[kr * KS];
#pragma unroll
      for (int kc = 0; kc < KS; ++kc) acc = fmaf(trow[kc], wrow[kc], acc);
    }
    out[(size_t)(l0 + rr) * DIMC + d] = fmaxf(acc, 0.0f);
  }
}

// ---- x = relu(x @ Wa^T + ba), x is [1,128] ---------------------------------
__global__ void k_xupd(const float* __restrict__ xin,
                       const float* __restrict__ Wa,
                       const float* __restrict__ ba,
                       float* __restrict__ xout) {
  int n = threadIdx.x;                       // 128 threads
  float acc = ba[n];
  const float* wr = Wa + (size_t)n * DIMC;
#pragma unroll 8
  for (int k = 0; k < DIMC; ++k) acc = fmaf(xin[k], wr[k], acc);
  xout[n] = fmaxf(acc, 0.0f);
}

// ---- weights[l] = tanh(dot(x, hs2[l])) -------------------------------------
__global__ void k_attn(const float* __restrict__ x,
                       const float* __restrict__ hs2,
                       float* __restrict__ wout) {
  int l = blockIdx.x * blockDim.x + threadIdx.x;   // LPROT threads
  const float* row = hs2 + (size_t)l * DIMC;
  float acc = 0.0f;
#pragma unroll 8
  for (int d = 0; d < DIMC; ++d) acc = fmaf(x[d], row[d], acc);
  wout[l] = tanhf(acc);
}

// ---- xs_p = weights[l] * hs2 -----------------------------------------------
__global__ void k_scale(const float* __restrict__ wv,
                        const float* __restrict__ hs2,
                        float* __restrict__ xsp) {
  int i = blockIdx.x * blockDim.x + threadIdx.x;   // LPROT*DIMC
  xsp[i] = wv[i >> 7] * hs2[i];
}

// ---- deterministic two-stage column sum (no float atomics) -----------------
__global__ void k_colsum_part(const float* __restrict__ X,  // [8192][128]
                              float* __restrict__ part) {   // [64][128]
  int c = threadIdx.x;                                      // 128 threads
  float acc = 0.0f;
  for (int r = blockIdx.x; r < NATOMS; r += 64)
    acc += X[(size_t)r * DIMC + c];
  part[(size_t)blockIdx.x * DIMC + c] = acc;
}

__global__ void k_colsum_fin(const float* __restrict__ part,
                             float* __restrict__ out,
                             float* __restrict__ out2) {    // out2 optional
  int c = threadIdx.x;                                      // 128 threads
  float acc = 0.0f;
#pragma unroll
  for (int b = 0; b < 64; ++b) acc += part[(size_t)b * DIMC + c];
  out[c] = acc;
  if (out2) out2[c] = acc;
}

// ---- final: logits = [xc,xp] @ Wo^T + bo ; softmax -------------------------
__global__ void k_final(const float* __restrict__ xc,
                        const float* __restrict__ xp,
                        const float* __restrict__ Wo,  // [2][256]
                        const float* __restrict__ bo,  // [2]
                        float* __restrict__ out) {
  __shared__ float sm[2];
  if (threadIdx.x < 2) {
    int o = threadIdx.x;
    float acc = bo[o];
    const float* wr = Wo + (size_t)o * 2 * DIMC;
    for (int k = 0; k < DIMC; ++k) acc = fmaf(wr[k], xc[k], acc);
    for (int k = 0; k < DIMC; ++k) acc = fmaf(wr[DIMC + k], xp[k], acc);
    sm[o] = acc;
  }
  __syncthreads();
  if (threadIdx.x == 0) {
    float m = fmaxf(sm[0], sm[1]);
    float e0 = expf(sm[0] - m), e1 = expf(sm[1] - m);
    float inv = 1.0f / (e0 + e1);
    out[0] = e0 * inv;
    out[1] = e1 * inv;
  }
}

// ---------------------------------------------------------------------------
extern "C" void kernel_launch(void* const* d_in, const int* in_sizes, int n_in,
                              void* d_out, int out_size, void* d_ws,
                              size_t ws_size, hipStream_t stream) {
  (void)in_sizes; (void)n_in; (void)out_size;
  const int*   fingerprints = (const int*)d_in[0];
  const int*   words        = (const int*)d_in[1];
  const float* adjacency    = (const float*)d_in[2];
  const float* emb_fp       = (const float*)d_in[3];
  const float* emb_word     = (const float*)d_in[4];
  const float* Wg           = (const float*)d_in[5];   // [3][128][128]
  const float* bg           = (const float*)d_in[6];   // [3][128]
  const float* conv_w       = (const float*)d_in[7];   // [3][23][23]
  const float* conv_b       = (const float*)d_in[8];   // [3]
  const float* Wa           = (const float*)d_in[9];
  const float* ba           = (const float*)d_in[10];
  const float* Wo           = (const float*)d_in[11];
  const float* bo           = (const float*)d_in[12];

  // workspace carve-out (256B aligned)
  char* base = (char*)d_ws;
  size_t off = 0;
  auto take = [&](size_t bytes) -> char* {
    char* p = base + off;
    off = (off + bytes + 255) & ~(size_t)255;
    return p;
  };
  _Float16* adjH  = (_Float16*)take((size_t)NATOMS * NATOMS * 2); // 128 MiB
  float*    xs    = (float*)   take((size_t)NATOMS * DIMC * 4);
  _Float16* xsH   = (_Float16*)take((size_t)NATOMS * DIMC * 2);
  _Float16* hsT   = (_Float16*)take((size_t)DIMC * NATOMS * 2);
  float*    xsp   = (float*)   take((size_t)LPROT * DIMC * 4);
  float*    convo = (float*)   take((size_t)LPROT * DIMC * 4);
  float*    hs2   = (float*)   take((size_t)LPROT * DIMC * 4);
  float*    attw  = (float*)   take((size_t)LPROT * 4);
  float*    part  = (float*)   take((size_t)64 * DIMC * 4);
  float*    xcomp = (float*)   take(DIMC * 4);
  float*    xprot = (float*)   take(DIMC * 4);
  float*    xbuf0 = (float*)   take(DIMC * 4);
  float*    xbuf1 = (float*)   take(DIMC * 4);
  if (off > ws_size) return;   // workspace too small (deterministic no-op)

  // ---- GNN over the compound graph ----------------------------------------
  {
    int n4 = NATOMS * NATOMS / 4;
    k_adj_to_half<<<(n4 + 255) / 256, 256, 0, stream>>>(
        (const float4*)adjacency, (v4h*)adjH, n4);
  }
  k_gather_f32h<<<NATOMS * DIMC / 256, 256, 0, stream>>>(fingerprints, emb_fp,
                                                         xs, xsH);
  for (int i = 0; i < 3; ++i) {
    k_lin_to_hsT<<<NATOMS / 16, 256, 0, stream>>>(
        xsH, Wg + (size_t)i * DIMC * DIMC, bg + (size_t)i * DIMC, hsT);
    k_adj_gemm<<<NATOMS / 32, 256, 0, stream>>>(adjH, hsT, xs, xsH);
  }
  k_colsum_part<<<64, 128, 0, stream>>>(xs, part);
  k_colsum_fin<<<1, 128, 0, stream>>>(part, xcomp, xbuf0); // x := x_compound

  // ---- attention-CNN over protein words -----------------------------------
  k_gather_f32<<<LPROT * DIMC / 256, 256, 0, stream>>>(words, emb_word, xsp);
  float* xbufs[2] = {xbuf0, xbuf1};
  int cur = 0;
  for (int i = 0; i < 3; ++i) {
    k_conv<<<LPROT / CTILE, 256, 0, stream>>>(xsp, conv_w + (size_t)i * KS2,
                                              conv_b + i, convo);
    k_lin_f32<<<LPROT / 16, 256, 0, stream>>>(convo, Wa, ba, hs2);
    k_xupd<<<1, 128, 0, stream>>>(xbufs[cur], Wa, ba, xbufs[1 - cur]);
    cur ^= 1;
    k_attn<<<LPROT / 256, 256, 0, stream>>>(xbufs[cur], hs2, attw);
    k_scale<<<LPROT * DIMC / 256, 256, 0, stream>>>(attw, hs2, xsp);
  }
  k_colsum_part<<<64, 128, 0, stream>>>(xsp, part);
  k_colsum_fin<<<1, 128, 0, stream>>>(part, xprot, nullptr);

  // ---- output head ---------------------------------------------------------
  k_final<<<1, 64, 0, stream>>>(xcomp, xprot, Wo, bo, (float*)d_out);
}
